// FFM_model_46591805227090
// MI455X (gfx1250) — matrix operand order repaired and checked by reading, not compiled, not run on hardware
//
#include <hip/hip_runtime.h>

typedef __attribute__((ext_vector_type(16))) _Float16 v16h;
typedef __attribute__((ext_vector_type(8)))  float    v8f;

static constexpr int NN      = 256;   // feature dim
static constexpr int FF      = 32;    // fields
static constexpr int KK      = 8;     // latent dim
static constexpr int MB      = 128;   // batch rows per workgroup
static constexpr int THREADS = 256;   // 8 waves (wave32)

// ---------------------------------------------------------------------------
// Kernel 1: Wu[i][j] = (i<j) ? sum_k v[i, f_j, k] * v[j, f_i, k] : 0   (f16)
// Stored row-major over i: each row holds the B fragments this kernel loads for
// Y = X @ Wu^T  (N index = i across lanes, K index = j contiguous per lane).
// ---------------------------------------------------------------------------
__global__ __launch_bounds__(NN)
void build_wu_f16(const float* __restrict__ v,
                  const int*   __restrict__ fidx,
                  _Float16*    __restrict__ w16) {
    const int i  = blockIdx.x;
    const int j  = threadIdx.x;
    const int fi = fidx[i];
    const int fj = fidx[j];
    const float* vi = v + (i * FF + fj) * KK;
    const float* vj = v + (j * FF + fi) * KK;
    float s = 0.f;
#pragma unroll
    for (int k = 0; k < KK; ++k) s += vi[k] * vj[k];
    w16[i * NN + j] = (i < j) ? (_Float16)s : (_Float16)0.f;
}

// ---------------------------------------------------------------------------
// Kernel 2: fused  Y = X @ Wu^T  (f16 WMMA, f32 accum)  +
//           out[b] = b_lin + sum_i x[b,i] * (w_lin[i] + Y[b,i])
// ---------------------------------------------------------------------------
__global__ __launch_bounds__(THREADS)
void ffm_fused(const float*    __restrict__ x,
               const float*    __restrict__ w_lin,
               const float*    __restrict__ b_lin,
               const _Float16* __restrict__ w16,
               float*          __restrict__ out) {
    extern __shared__ char smem[];
    _Float16* sW = (_Float16*)smem;                              // 256*256 f16 = 128 KB
    float*    sX = (float*)(smem + NN * NN * sizeof(_Float16));  // 128*256 f32 = 128 KB

    const int tid = threadIdx.x;
    const int blk = blockIdx.x;

    // ---- cooperative LDS fill (16B/lane vector traffic) ----
    {
        const float4* wg  = (const float4*)w16;
        float4*       sw4 = (float4*)sW;
#pragma unroll
        for (int t = 0; t < (NN * NN * 2 / 16) / THREADS; ++t)   // 32 iters
            sw4[tid + t * THREADS] = wg[tid + t * THREADS];
        const float4* xg  = (const float4*)(x + (size_t)blk * MB * NN);
        float4*       sx4 = (float4*)sX;
#pragma unroll
        for (int t = 0; t < (MB * NN * 4 / 16) / THREADS; ++t)   // 32 iters
            sx4[tid + t * THREADS] = xg[tid + t * THREADS];
    }
    __syncthreads();

    const int lane = tid & 31;
    const int wv   = tid >> 5;    // wave id == row tile (0..7)
    const int hs   = lane >> 4;   // lane-half select
    const int ml   = lane & 15;

    // ---- build the 8 A fragments (16x32 f16) for this wave's row tile ----
    // Layout: lanes 0-15 hold row M=lane with K = kb*32 + {0..15};
    //         lanes 16-31 hold row M=lane-16 with K = kb*32 + {16..31}.
    const float* xrow = sX + (wv * 16 + ml) * NN;
    v16h a[8];
#pragma unroll
    for (int kb = 0; kb < 8; ++kb) {
        const int k0 = kb * 32 + hs * 16;
#pragma unroll
        for (int e = 0; e < 8; ++e) a[kb][e]     = (_Float16)xrow[k0 + e];
#pragma unroll
        for (int e = 0; e < 8; ++e) a[kb][e + 8] = (_Float16)xrow[k0 + 8 + e];
    }

    float acc[8];
#pragma unroll
    for (int r = 0; r < 8; ++r) acc[r] = 0.f;

#pragma unroll
    for (int nt = 0; nt < 16; ++nt) {
        // Wu is strictly upper-triangular: tile (N cols i = nt*16.., K rows j = kb*32..)
        // is all-zero unless some i<j exists, i.e. 16*nt < 32*kb+31.
        const int kbmin = (16 * nt + 1) >> 5;
        v8f c = {};
#pragma unroll
        for (int kb = 0; kb < 8; ++kb) {
            if (kb < kbmin) continue;   // constant-folded after unroll
            // B fragment (32x16 f16): lane = N col, K = kb*32 + hs*16 + e, contiguous 32B
            const v16h b = *(const v16h*)(sW + (nt * 16 + ml) * NN + kb * 32 + hs * 16);
            c = __builtin_amdgcn_wmma_f32_16x16x32_f16(
                    false, a[kb], false, b, (short)0, c, false, false);
        }
        // C layout: lanes 0-15 -> rows wv*16 + r, lanes 16-31 -> rows wv*16 + 8 + r;
        // N col = nt*16 + ml.  Fold in w_lin here (each n visited exactly once).
        const int   n  = nt * 16 + ml;
        const float wl = w_lin[n];
#pragma unroll
        for (int r = 0; r < 8; ++r) {
            const float xv = sX[(wv * 16 + hs * 8 + r) * NN + n];
            acc[r] += xv * (wl + c[r]);
        }
    }

    // ---- reduce over the 16 lanes of each half (rows are per-half, wave32) ----
#pragma unroll
    for (int m = 1; m < 16; m <<= 1) {
#pragma unroll
        for (int r = 0; r < 8; ++r) acc[r] += __shfl_xor(acc[r], m, 32);
    }

    if (ml == 0) {
        const float bl = b_lin[0];
#pragma unroll
        for (int r = 0; r < 8; ++r)
            out[(size_t)blk * MB + wv * 16 + hs * 8 + r] = acc[r] + bl;
    }
}

// ---------------------------------------------------------------------------
extern "C" void kernel_launch(void* const* d_in, const int* in_sizes, int n_in,
                              void* d_out, int out_size, void* d_ws, size_t ws_size,
                              hipStream_t stream) {
    const float* x     = (const float*)d_in[0];
    const float* w_lin = (const float*)d_in[1];
    const float* b_lin = (const float*)d_in[2];
    const float* v     = (const float*)d_in[3];
    const int*   fidx  = (const int*)d_in[4];
    float*       out   = (float*)d_out;
    _Float16*    w16   = (_Float16*)d_ws;     // 256*256 f16 = 128 KB scratch

    const int Bsz = in_sizes[0] / NN;         // 65536

    build_wu_f16<<<NN, NN, 0, stream>>>(v, fidx, w16);

    const size_t shmem = (size_t)NN * NN * sizeof(_Float16)
                       + (size_t)MB * NN * sizeof(float);   // 256 KB (<=320 KB/WG)
    ffm_fused<<<Bsz / MB, THREADS, shmem, stream>>>(x, w_lin, b_lin, w16, out);
}
